// GCNConv_7602092114484
// MI455X (gfx1250) — matrix operand order repaired and checked by reading, not compiled, not run on hardware
//
#include <hip/hip_runtime.h>

// ---- POD vector types (ext_vector unions are safe; HIP's uint4 is not) ----
typedef __bf16        bf16_t;
typedef bf16_t        v16bf __attribute__((ext_vector_type(16)));
typedef float         v8f   __attribute__((ext_vector_type(8)));
typedef float         v4f   __attribute__((ext_vector_type(4)));
typedef unsigned int  u32x4 __attribute__((ext_vector_type(4)));

#define N_NODES 16384
#define DIMS    256

// Generic (flat) pointer to a __shared__ object -> wave-relative LDS byte offset.
// On amdgcn, flat LDS addresses carry the LDS offset in the low 32 bits.
__device__ __forceinline__ unsigned lds_offset(const void* p) {
    return (unsigned)(unsigned long long)p;
}

// CDNA5 async copy: global -> LDS, 16 bytes per lane, tracked by ASYNCcnt.
__device__ __forceinline__ void async_load_to_lds_b128(unsigned ldsOff, const void* gaddr) {
    asm volatile("global_load_async_to_lds_b128 %0, %1, off"
                 :: "v"(ldsOff), "v"(gaddr)
                 : "memory");
}
__device__ __forceinline__ void wait_asynccnt0() {
    asm volatile("s_wait_asynccnt 0x0" ::: "memory");
}

// =====================================================================
// Stage 1: XWT[n][i] = sum_k X[i][k] * W[k][n]   (bf16 output, [256,16384])
// Computed as C = (W^T) @ (X^T): A-frag = W^T rows (from LDS), B-frag = X rows
// (contiguous per lane), D stores land contiguous in XWT rows.
// Grid: (64 i-blocks of 256, 8 n-blocks of 32). Block: 256 threads (8 waves).
// =====================================================================
__global__ __launch_bounds__(256) void gcn_stage1_xwt(
    const float* __restrict__ X,   // [16384,256]
    const float* __restrict__ W,   // [256,256]
    bf16_t* __restrict__ XWT)      // [256,16384]
{
    __shared__ __align__(16) unsigned char ldsW[32 * 528]; // W^T bf16, 32 n-rows, 256 k + pad

    const int t    = threadIdx.x;
    const int lane = t & 31;
    const int wv   = t >> 5;
    const int hi   = lane >> 4;
    const int ln   = lane & 15;
    const int i0   = blockIdx.x * 256;
    const int n0   = blockIdx.y * 32;

    // Cooperative stage of W^T (bf16) into LDS
    {
        const int n  = t & 31;
        const int kc = (t >> 5) * 32;
        union { bf16_t h[32]; u32x4 q[4]; } wb;
        #pragma unroll
        for (int k = 0; k < 32; ++k)
            wb.h[k] = (bf16_t)W[(size_t)(kc + k) * DIMS + (n0 + n)];
        u32x4* d = (u32x4*)(ldsW + n * 528 + kc * 2);
        #pragma unroll
        for (int q = 0; q < 4; ++q) d[q] = wb.q[q];
    }
    __syncthreads();

    const int ng    = wv & 1;
    const int ig    = wv >> 1;
    const int naL   = ng * 16;
    const int naG   = n0 + ng * 16;
    const int ibase = i0 + ig * 64;

    v8f acc[4];
    #pragma unroll
    for (int it = 0; it < 4; ++it)
        acc[it] = (v8f){0.f,0.f,0.f,0.f,0.f,0.f,0.f,0.f};

    #pragma unroll
    for (int kb = 0; kb < DIMS / 32; ++kb) {
        union { u32x4 u[2]; v16bf v; } af;
        const unsigned char* ap = ldsW + (naL + ln) * 528 + (kb * 32 + hi * 16) * 2;
        af.u[0] = *(const u32x4*)(ap);
        af.u[1] = *(const u32x4*)(ap + 16);

        #pragma unroll
        for (int it = 0; it < 4; ++it) {
            const float* xp = X + (size_t)(ibase + it * 16 + ln) * DIMS + kb * 32 + hi * 16;
            v4f b0 = *(const v4f*)(xp);
            v4f b1 = *(const v4f*)(xp + 4);
            v4f b2 = *(const v4f*)(xp + 8);
            v4f b3 = *(const v4f*)(xp + 12);
            union { bf16_t h[16]; v16bf v; } bfv;
            #pragma unroll
            for (int e = 0; e < 4; ++e) {
                bfv.h[e]      = (bf16_t)b0[e];
                bfv.h[4 + e]  = (bf16_t)b1[e];
                bfv.h[8 + e]  = (bf16_t)b2[e];
                bfv.h[12 + e] = (bf16_t)b3[e];
            }
            acc[it] = __builtin_amdgcn_wmma_f32_16x16x32_bf16(
                false, af.v, false, bfv.v, (short)0, acc[it], false, false);
        }
    }

    #pragma unroll
    for (int it = 0; it < 4; ++it) {
        #pragma unroll
        for (int r = 0; r < 8; ++r) {
            XWT[(size_t)(naG + r + hi * 8) * (size_t)N_NODES + (ibase + it * 16 + ln)] =
                (bf16_t)acc[it][r];
        }
    }
}

// =====================================================================
// Stage 2: Y = ReLU(A @ XW).  A fp32 streamed once (1 GiB) and converted to
// bf16 in registers; XWT (bf16, L2-resident) staged via async global->LDS
// copies into a double-buffered 72 KB LDS tile (rows padded to 144 B).
// Grid: 256 WGs (64 rows each). Block: 256 threads = 8 waves.
// Wave: 16 M-rows x 128 N-cols; K-loop 256 x 64 (16 WMMAs / iteration).
// =====================================================================
__global__ __launch_bounds__(256) void gcn_stage2_agg(
    const float* __restrict__ A,     // [16384,16384]
    const bf16_t* __restrict__ XWT,  // [256,16384]
    float* __restrict__ Y)           // [16384,256]
{
    __shared__ __align__(16) unsigned char ldsB[2][256 * 144]; // 72 KB

    const int t    = threadIdx.x;
    const int lane = t & 31;
    const int wv   = t >> 5;
    const int hi   = lane >> 4;
    const int ln   = lane & 15;
    const int mg   = wv >> 1;               // 4 M-groups of 16 rows
    const int nh   = wv & 1;                // 2 N-halves of 128
    const int rowBase = blockIdx.x * 64 + mg * 16;
    const int myRow   = rowBase + ln;
    const int nbase   = nh * 128;

    // B staging: thread t owns XWT row n=t, async-copies 128 B per 64-wide k-block
    const char* srcB = (const char*)(XWT + (size_t)t * (size_t)N_NODES);
    const unsigned ldsRow0 = lds_offset(&ldsB[0][t * 144]);
    const unsigned ldsRow1 = lds_offset(&ldsB[1][t * 144]);

    // A: lane reads its private row; chunks match the 16-bit A-fragment layout
    const float* aRow = A + (size_t)myRow * (size_t)N_NODES + hi * 8;

    v8f acc[8];
    #pragma unroll
    for (int nt = 0; nt < 8; ++nt)
        acc[nt] = (v8f){0.f,0.f,0.f,0.f,0.f,0.f,0.f,0.f};

    // Prologue: async-stage k-block 0 into buffer 0
    #pragma unroll
    for (int j = 0; j < 8; ++j)
        async_load_to_lds_b128(ldsRow0 + j * 16, srcB + j * 16);
    wait_asynccnt0();
    __syncthreads();

    const int NKB = N_NODES / 64;   // 256
    int cur = 0;
    for (int kb = 0; kb < NKB; ++kb) {
        // Kick off async staging of the next B tile into the alternate buffer;
        // the 16 WMMAs below form its latency shadow.
        if (kb + 1 < NKB) {
            const char* s = srcB + (size_t)(kb + 1) * 128;
            const unsigned d = cur ? ldsRow0 : ldsRow1;
            #pragma unroll
            for (int j = 0; j < 8; ++j)
                async_load_to_lds_b128(d + j * 16, s + j * 16);
        }

        const unsigned char* bufc = cur ? ldsB[1] : ldsB[0];
        #pragma unroll
        for (int ks = 0; ks < 2; ++ks) {
            // A fragment: fp32 load -> bf16 convert in regs
            const float* ap = aRow + (size_t)kb * 64 + ks * 32;
            v4f a0 = *(const v4f*)(ap);        // K base + 0..3
            v4f a1 = *(const v4f*)(ap + 4);    // + 4..7
            v4f a2 = *(const v4f*)(ap + 16);   // + 16..19
            v4f a3 = *(const v4f*)(ap + 20);   // + 20..23
            union { bf16_t h[16]; v16bf v; } af;
            #pragma unroll
            for (int e = 0; e < 4; ++e) {
                af.h[e]      = (bf16_t)a0[e];
                af.h[4 + e]  = (bf16_t)a1[e];
                af.h[8 + e]  = (bf16_t)a2[e];
                af.h[12 + e] = (bf16_t)a3[e];
            }

            const unsigned char* bbase = bufc + (nbase + ln) * 144 + ks * 64 + hi * 16;
            #pragma unroll
            for (int nt = 0; nt < 8; ++nt) {
                union { u32x4 u[2]; v16bf v; } bfv;
                const u32x4* bp = (const u32x4*)(bbase + nt * 16 * 144);
                bfv.u[0] = bp[0];
                bfv.u[1] = bp[2];
                acc[nt] = __builtin_amdgcn_wmma_f32_16x16x32_bf16(
                    false, af.v, false, bfv.v, (short)0, acc[nt], false, false);
            }
        }

        // Make the freshly staged tile visible to all waves
        wait_asynccnt0();
        __syncthreads();
        cur ^= 1;
    }

    // Fused ReLU + store: D element (r) -> row rowBase + r + 8*hi, col nbase+nt*16+ln
    float* yb = Y + (size_t)(rowBase + hi * 8) * DIMS + nbase + ln;
    #pragma unroll
    for (int nt = 0; nt < 8; ++nt) {
        #pragma unroll
        for (int r = 0; r < 8; ++r) {
            float v = acc[nt][r];
            yb[(size_t)r * DIMS + nt * 16] = v > 0.f ? v : 0.f;
        }
    }
}

// =====================================================================
extern "C" void kernel_launch(void* const* d_in, const int* in_sizes, int n_in,
                              void* d_out, int out_size, void* d_ws, size_t ws_size,
                              hipStream_t stream) {
    const float* X = (const float*)d_in[0];   // [16384,256]
    const float* A = (const float*)d_in[1];   // [16384,16384]
    const float* W = (const float*)d_in[2];   // [256,256]
    float*       Y = (float*)d_out;           // [16384,256]
    bf16_t*    XWT = (bf16_t*)d_ws;           // [256,16384] bf16 = 8 MB scratch

    // Stage 1: XWT = (X @ W)^T in bf16
    gcn_stage1_xwt<<<dim3(N_NODES / 256, DIMS / 32), 256, 0, stream>>>(X, W, XWT);
    // Stage 2: Y = ReLU(A @ XW)
    gcn_stage2_agg<<<dim3(N_NODES / 64), 256, 0, stream>>>(A, XWT, Y);
}
